// Model_63428077027896
// MI455X (gfx1250) — matrix operand order repaired and checked by reading, not compiled, not run on hardware
//
#include <hip/hip_runtime.h>

// ---------------------------------------------------------------------------
// Types
// ---------------------------------------------------------------------------
typedef __attribute__((ext_vector_type(16))) __bf16        v16bf;
typedef __attribute__((ext_vector_type(8)))  float         v8f;
typedef __attribute__((ext_vector_type(4)))  unsigned int  u32x4;
typedef __attribute__((ext_vector_type(4)))  float         f32x4;

#define LP 40   // LDS row pitch in bf16 elements (32 data + 8 pad)

__device__ __forceinline__ unsigned short f2bf(float x) {
  unsigned int u = __float_as_uint(x);
  u += 0x7FFFu + ((u >> 16) & 1u);           // round-to-nearest-even
  return (unsigned short)(u >> 16);
}

// ---------------------------------------------------------------------------
// Elementwise f32 -> bf16 (weight staging)
// ---------------------------------------------------------------------------
__global__ __launch_bounds__(256)
void f32_to_bf16_kernel(const float* __restrict__ in,
                        unsigned short* __restrict__ out, long long n) {
  long long stride = (long long)gridDim.x * blockDim.x;
  for (long long i = (long long)blockIdx.x * blockDim.x + threadIdx.x; i < n; i += stride)
    out[i] = f2bf(in[i]);
}

// ---------------------------------------------------------------------------
// Tiled transpose f32 -> bf16  (out[c][r] = in[r][c])
// ---------------------------------------------------------------------------
__global__ __launch_bounds__(256)
void transpose_f32_bf16(const float* __restrict__ in,
                        unsigned short* __restrict__ out,
                        int R, int Ccols, long long sInZ, long long sOutZ) {
  __shared__ unsigned short t[32][33];
  in  += (long long)blockIdx.z * sInZ;
  out += (long long)blockIdx.z * sOutZ;
  const int c0 = blockIdx.x * 32;
  const int r0 = blockIdx.y * 32;
  for (int i = threadIdx.y; i < 32; i += 8)
    t[i][threadIdx.x] = f2bf(in[(size_t)(r0 + i) * Ccols + c0 + threadIdx.x]);
  __syncthreads();
  for (int i = threadIdx.y; i < 32; i += 8)
    out[(size_t)(c0 + i) * R + r0 + threadIdx.x] = t[threadIdx.x][i];
}

// ---------------------------------------------------------------------------
// Tiled transpose bf16 -> bf16
// ---------------------------------------------------------------------------
__global__ __launch_bounds__(256)
void transpose_bf16(const unsigned short* __restrict__ in,
                    unsigned short* __restrict__ out,
                    int R, int Ccols, long long sInZ, long long sOutZ) {
  __shared__ unsigned short t[32][33];
  in  += (long long)blockIdx.z * sInZ;
  out += (long long)blockIdx.z * sOutZ;
  const int c0 = blockIdx.x * 32;
  const int r0 = blockIdx.y * 32;
  for (int i = threadIdx.y; i < 32; i += 8)
    t[i][threadIdx.x] = in[(size_t)(r0 + i) * Ccols + c0 + threadIdx.x];
  __syncthreads();
  for (int i = threadIdx.y; i < 32; i += 8)
    out[(size_t)(c0 + i) * R + r0 + threadIdx.x] = t[threadIdx.x][i];
}

// ---------------------------------------------------------------------------
// NT GEMM:  D[m,n] = scale * sum_k A[m,k]*B[n,k]  (+ bias[m] or bias[n])
// A: Mdim x Kdim row-major bf16 (lda), B: Ndim x Kdim row-major bf16 (ldb).
// 128x128 block tile, BK=32, 256 threads = 8 wave32 (2x4), 64x32 per wave.
// Uses v_wmma_f32_16x16x32_bf16; fragments per CDNA5 ISA VGPR layouts.
// ---------------------------------------------------------------------------
__global__ __launch_bounds__(256)
void gemm_nt_bf16(const unsigned short* __restrict__ A, int lda, long long sAz,
                  const unsigned short* __restrict__ Bm, int ldb, long long sBz,
                  void* __restrict__ Dv, int ldd, long long sDz1, long long sDz2, int zmod,
                  int Mdim, int Ndim, int Kdim,
                  const float* __restrict__ bias, int biasMode,   // 0 none,1 row,2 col
                  float scale, int outBf16) {
  (void)Mdim; (void)Ndim;
  __shared__ unsigned short lA[128 * LP];
  __shared__ unsigned short lB[128 * LP];

  const int z = blockIdx.z;
  A  += (long long)z * sAz;
  Bm += (long long)z * sBz;
  const long long doff = (long long)(z / zmod) * sDz1 + (long long)(z % zmod) * sDz2;

  const int m0 = blockIdx.y * 128;
  const int n0 = blockIdx.x * 128;

  const int tid  = threadIdx.x;
  const int lane = tid & 31;
  const int wave = tid >> 5;
  const int wm   = (wave >> 2) * 64;     // wave row offset (2 rows of waves)
  const int wn   = (wave & 3) * 32;      // wave col offset (4 cols of waves)

  const int lrow = tid >> 1;             // 0..127: tile row loaded by this thread
  const int lcol = (tid & 1) << 4;       // 0 or 16: k-offset loaded by this thread

  const int r15 = lane & 15;
  const int kbA = (lane >> 4) << 3;      // A frag k-base: 0 or 8
  const int kbB = (lane >> 4) << 3;      // B frag k-base: 0 or 8

  v8f acc[4][2] = {};

  const unsigned short* gA = A  + (size_t)(m0 + lrow) * lda + lcol;
  const unsigned short* gB = Bm + (size_t)(n0 + lrow) * ldb + lcol;

  for (int k0 = 0; k0 < Kdim; k0 += 32) {
    u32x4 a0 = ((const u32x4*)(gA + k0))[0];
    u32x4 a1 = ((const u32x4*)(gA + k0))[1];
    u32x4 b0 = ((const u32x4*)(gB + k0))[0];
    u32x4 b1 = ((const u32x4*)(gB + k0))[1];
    if (k0 + 32 < Kdim) {                      // prefetch next K tile
      __builtin_prefetch(gA + k0 + 32, 0, 1);
      __builtin_prefetch(gB + k0 + 32, 0, 1);
    }
    *(u32x4*)&lA[lrow * LP + lcol]     = a0;
    *(u32x4*)&lA[lrow * LP + lcol + 8] = a1;
    *(u32x4*)&lB[lrow * LP + lcol]     = b0;
    *(u32x4*)&lB[lrow * LP + lcol + 8] = b1;
    __syncthreads();

    // A fragment (16x32): lane<16 -> M=lane, K={0..7,16..23}; lane>=16 -> K={8..15,24..31}
    union { u32x4 u[2]; v16bf v; } fa[4], fb[2];
#pragma unroll
    for (int i = 0; i < 4; ++i) {
      const unsigned short* p = &lA[(wm + i * 16 + r15) * LP];
      fa[i].u[0] = *(const u32x4*)(p + kbA);
      fa[i].u[1] = *(const u32x4*)(p + kbA + 16);
    }
    // B fragment (32x16): lane<16 -> N=lane, K={0..7,16..23}; lane>=16 -> K={8..15,24..31}
#pragma unroll
    for (int j = 0; j < 2; ++j) {
      const unsigned short* p = &lB[(wn + j * 16 + r15) * LP];
      fb[j].u[0] = *(const u32x4*)(p + kbB);
      fb[j].u[1] = *(const u32x4*)(p + kbB + 16);
    }
#pragma unroll
    for (int i = 0; i < 4; ++i)
#pragma unroll
      for (int j = 0; j < 2; ++j)
        acc[i][j] = __builtin_amdgcn_wmma_f32_16x16x32_bf16(
            false, fa[i].v, false, fb[j].v, (short)0, acc[i][j], false, false);
    __syncthreads();
  }

  // Epilogue. C/D layout: VGPR r, lane L -> row = (L>>4)*8 + r, col = L&15.
  const int colB = n0 + wn + r15;
  const int rowB = m0 + wm + ((lane >> 4) << 3);
#pragma unroll
  for (int j = 0; j < 2; ++j) {
    const int col = colB + j * 16;
    const float cb = (biasMode == 2) ? bias[col] : 0.0f;
#pragma unroll
    for (int i = 0; i < 4; ++i) {
      const int rb = rowB + i * 16;
#pragma unroll
      for (int r = 0; r < 8; ++r) {
        const int row = rb + r;
        float v = acc[i][j][r] * scale;
        v += (biasMode == 1) ? bias[row] : cb;
        const long long idx = doff + (long long)row * ldd + col;
        if (outBf16) ((unsigned short*)Dv)[idx] = f2bf(v);
        else         ((float*)Dv)[idx]          = v;
      }
    }
  }
}

// ---------------------------------------------------------------------------
// Row softmax: 512-wide f32 rows -> bf16. One wave per row, 16 elems/lane.
// ---------------------------------------------------------------------------
__global__ __launch_bounds__(256)
void softmax_rows(const float* __restrict__ S, unsigned short* __restrict__ P) {
  const int row  = blockIdx.x * 8 + (threadIdx.x >> 5);
  const int lane = threadIdx.x & 31;
  const float* src = S + (size_t)row * 512 + lane * 16;
  f32x4 v[4];
#pragma unroll
  for (int i = 0; i < 4; ++i) v[i] = ((const f32x4*)src)[i];

  float m = -3.4e38f;
#pragma unroll
  for (int i = 0; i < 4; ++i)
#pragma unroll
    for (int c = 0; c < 4; ++c) m = fmaxf(m, v[i][c]);
#pragma unroll
  for (int off = 16; off > 0; off >>= 1) m = fmaxf(m, __shfl_xor(m, off, 32));

  float e[16]; float s = 0.0f;
#pragma unroll
  for (int i = 0; i < 4; ++i)
#pragma unroll
    for (int c = 0; c < 4; ++c) { float t = __expf(v[i][c] - m); e[i * 4 + c] = t; s += t; }
#pragma unroll
  for (int off = 16; off > 0; off >>= 1) s += __shfl_xor(s, off, 32);
  const float inv = 1.0f / s;

  unsigned int pk[8];
#pragma unroll
  for (int k = 0; k < 8; ++k)
    pk[k] = (unsigned int)f2bf(e[2 * k] * inv) |
            ((unsigned int)f2bf(e[2 * k + 1] * inv) << 16);
  u32x4* dst = (u32x4*)(P + (size_t)row * 512 + lane * 16);
  u32x4 o0 = { pk[0], pk[1], pk[2], pk[3] };
  u32x4 o1 = { pk[4], pk[5], pk[6], pk[7] };
  dst[0] = o0; dst[1] = o1;
}

// ---------------------------------------------------------------------------
// Host orchestration
// ---------------------------------------------------------------------------
extern "C" void kernel_launch(void* const* d_in, const int* in_sizes, int n_in,
                              void* d_out, int out_size, void* d_ws, size_t ws_size,
                              hipStream_t stream) {
  (void)in_sizes; (void)n_in; (void)out_size; (void)ws_size;
  const float* x    = (const float*)d_in[0];
  const float* Wqkv = (const float*)d_in[1];
  const float* bqkv = (const float*)d_in[2];
  const float* Wq   = (const float*)d_in[3];
  const float* bq   = (const float*)d_in[4];
  const float* Wk   = (const float*)d_in[5];
  const float* bk   = (const float*)d_in[6];
  const float* Wv   = (const float*)d_in[7];
  const float* bv   = (const float*)d_in[8];
  const float* Wout = (const float*)d_in[9];
  const float* bout = (const float*)d_in[10];

  char* ws = (char*)d_ws;
  // bf16 staging buffers (byte offsets)
  unsigned short* Wqkvh = (unsigned short*)(ws + 0LL);          // 12288x4096
  unsigned short* Wqh   = (unsigned short*)(ws + 100663296LL);  // 4096x4096
  unsigned short* Wkh   = (unsigned short*)(ws + 134217728LL);
  unsigned short* Wvh   = (unsigned short*)(ws + 167772160LL);
  unsigned short* Woh   = (unsigned short*)(ws + 201326592LL);
  unsigned short* Qh    = (unsigned short*)(ws + 234881024LL);  // 2 x 4096x4096
  unsigned short* Kh    = (unsigned short*)(ws + 301989888LL);
  unsigned short* Vh    = (unsigned short*)(ws + 369098752LL);
  unsigned short* QKVh  = (unsigned short*)(ws + 436207616LL);  // 2 x 12288x4096
  unsigned short* XT    = (unsigned short*)(ws + 637534208LL);  // 2 x 4096x4096 (x^T)
  // Aliases of dead regions (QKVh dead after projections; XT dead after qkv GEMM)
  float*          Sbuf  = (float*)(ws + 436207616LL);           // 16 x 512x512 f32
  unsigned short* Pbuf  = (unsigned short*)(ws + 452984832LL);  // 16 x 512x512 bf16
  unsigned short* OT    = (unsigned short*)(ws + 461373440LL);  // 2 x 4096x4096 bf16
  unsigned short* VhT   = XT;                                   // 16 x 4096x512

  // 1) weights -> bf16
  f32_to_bf16_kernel<<<4096, 256, 0, stream>>>(Wqkv, Wqkvh, 50331648LL);
  f32_to_bf16_kernel<<<2048, 256, 0, stream>>>(Wq,   Wqh,   16777216LL);
  f32_to_bf16_kernel<<<2048, 256, 0, stream>>>(Wk,   Wkh,   16777216LL);
  f32_to_bf16_kernel<<<2048, 256, 0, stream>>>(Wv,   Wvh,   16777216LL);
  f32_to_bf16_kernel<<<2048, 256, 0, stream>>>(Wout, Woh,   16777216LL);
  // 2) x^T per batch: (4096c x 4096n) -> (n x c) bf16
  transpose_f32_bf16<<<dim3(128, 128, 2), dim3(32, 8), 0, stream>>>(
      x, XT, 4096, 4096, 16777216LL, 16777216LL);
  // 3) qkv[b][o][n] = Wqkv[o,:]·xT[n,:] + bqkv[o]   (M=12288,N=4096,K=4096)
  gemm_nt_bf16<<<dim3(32, 96, 2), 256, 0, stream>>>(
      Wqkvh, 4096, 0LL, XT, 4096, 16777216LL,
      QKVh, 4096, 50331648LL, 0LL, 1, 12288, 4096, 4096, bqkv, 1, 1.0f, 1);
  // 4) projections: D[r,m] = qkv_sel[r,:]·W[m,:] + b[m]
  gemm_nt_bf16<<<dim3(32, 32, 2), 256, 0, stream>>>(
      QKVh, 4096, 50331648LL, Wqh, 4096, 0LL,
      Qh, 4096, 16777216LL, 0LL, 1, 4096, 4096, 4096, bq, 2, 1.0f, 1);
  gemm_nt_bf16<<<dim3(32, 32, 2), 256, 0, stream>>>(
      QKVh + 16777216, 4096, 50331648LL, Wkh, 4096, 0LL,
      Kh, 4096, 16777216LL, 0LL, 1, 4096, 4096, 4096, bk, 2, 1.0f, 1);
  gemm_nt_bf16<<<dim3(32, 32, 2), 256, 0, stream>>>(
      QKVh + 33554432, 4096, 50331648LL, Wvh, 4096, 0LL,
      Vh, 4096, 16777216LL, 0LL, 1, 4096, 4096, 4096, bv, 2, 1.0f, 1);
  // 5) V^T per (b,h): (512d x 4096m) -> (m x d)
  transpose_bf16<<<dim3(128, 16, 16), dim3(32, 8), 0, stream>>>(
      Vh, VhT, 512, 4096, 2097152LL, 2097152LL);
  // 6) S[c,d] = 0.125 * Q[c,:]·K[d,:]   per (b,h): M=N=512, K=4096
  gemm_nt_bf16<<<dim3(4, 4, 16), 256, 0, stream>>>(
      Qh, 4096, 2097152LL, Kh, 4096, 2097152LL,
      Sbuf, 512, 262144LL, 0LL, 1, 512, 512, 4096, nullptr, 0, 0.125f, 0);
  // 7) softmax over d (8192 rows of 512)
  softmax_rows<<<1024, 256, 0, stream>>>(Sbuf, Pbuf);
  // 8) outT[m, h*512+c] = VhT[m,:]·P[c,:]   per (b,h): M=4096,N=512,K=512
  gemm_nt_bf16<<<dim3(4, 32, 16), 256, 0, stream>>>(
      VhT, 512, 2097152LL, Pbuf, 512, 262144LL,
      OT, 4096, 16777216LL, 512LL, 8, 4096, 512, 512, nullptr, 0, 1.0f, 1);
  // 9) y[b][o][n] = Wout[o,:]·outT[n,:] + bout[o]   (M=N=K=4096), f32 output
  gemm_nt_bf16<<<dim3(32, 32, 2), 256, 0, stream>>>(
      Woh, 4096, 0LL, OT, 4096, 16777216LL,
      d_out, 4096, 16777216LL, 0LL, 1, 4096, 4096, 4096, bout, 1, 1.0f, 0);
}
